// HypergraphDualChannel_78726750536358
// MI455X (gfx1250) — matrix operand order repaired and checked by reading, not compiled, not run on hardware
//
#include <hip/hip_runtime.h>

#define QN 16384
#define SN 16384
#define FN 64
#define HN 32
#define EN 393216

typedef __attribute__((ext_vector_type(16))) __bf16 v16bf;
typedef __attribute__((ext_vector_type(8)))  float  v8f;

#define WMMA_BF16(a,b,c) \
  __builtin_amdgcn_wmma_f32_16x16x32_bf16(false,(a),false,(b),(short)0,(c),false,false)

struct uint4x2 { uint4 a, b; };
__device__ __forceinline__ v16bf make_v16bf(uint4 lo, uint4 hi){
  uint4x2 t{lo, hi};
  return __builtin_bit_cast(v16bf, t);
}

__device__ __forceinline__ float bflo(unsigned u){ return __builtin_bit_cast(float, u << 16); }
__device__ __forceinline__ float bfhi(unsigned u){ return __builtin_bit_cast(float, u & 0xffff0000u); }
__device__ __forceinline__ unsigned short bfbits(float f){
  return __builtin_bit_cast(unsigned short, (__bf16)f);
}

// order-preserving float<->uint encoding for atomicMax-based segment max
__device__ __forceinline__ unsigned fenc(float f){
  unsigned b = __builtin_bit_cast(unsigned, f);
  return (b & 0x80000000u) ? ~b : (b | 0x80000000u);
}
__device__ __forceinline__ float fdec(unsigned u){
  unsigned b = (u & 0x80000000u) ? (u & 0x7fffffffu) : ~u;
  return __builtin_bit_cast(float, b);
}

// A fragment: 16x32 bf16 from row-major LDS rows (leading dim ld, elems).
// ISA layout: lanes 0-15 row m, elems 0..7 = K kb+hl*8+e, elems 8..15 = K kb+16+hl*8+e.
// Both 8-elem groups are contiguous & 16B aligned -> two ds_load_b128.
__device__ __forceinline__ v16bf frag_A(const __bf16* rows, int ld, int kb, int lane){
  const int hl = (lane >> 4) & 1, m = lane & 15;
  const __bf16* r = rows + m * ld + kb + hl*8;
  uint4 lo = *reinterpret_cast<const uint4*>(r);
  uint4 hi = *reinterpret_cast<const uint4*>(r + 16);
  return make_v16bf(lo, hi);
}
// B fragment from swizzled weights: element order [(kb*4+nb)*32+lane][16] contiguous
__device__ __forceinline__ v16bf frag_B(const __bf16* Wsw, int kb, int nb, int lane){
  const __bf16* p = Wsw + ((((kb*4 + nb) << 5) + lane) << 4);
  uint4 lo = reinterpret_cast<const uint4*>(p)[0];
  uint4 hi = reinterpret_cast<const uint4*>(p)[1];
  return make_v16bf(lo, hi);
}

// ---------------- utility kernels ----------------
__global__ __launch_bounds__(256) void k_cvt_bf16(const float* __restrict__ s,
                                                  __bf16* __restrict__ d, int n){
  int i = blockIdx.x * 256 + threadIdx.x;
  if (i < n) d[i] = (__bf16)s[i];
}
// swizzle row-major W[K][64] f32 -> bf16 fragment-major [(kb*4+nb)*32+lane][16]
__global__ __launch_bounds__(256) void k_swz_w(const float* __restrict__ W,
                                               __bf16* __restrict__ out, int K){
  int i = blockIdx.x * 256 + threadIdx.x;
  if (i >= K * 64) return;
  int e    = i & 15;
  int lane = (i >> 4) & 31;
  int t    = i >> 9;            // kb*4 + nb
  int nb = t & 3, kb = t >> 2;
  int k = kb*32 + (e >> 3)*16 + ((lane >> 4) & 1)*8 + (e & 7);
  int n = nb*16 + (lane & 15);
  out[i] = (__bf16)W[k*64 + n];
}
__global__ __launch_bounds__(256) void k_zero_f32(float* p, int n){
  int i = blockIdx.x * 256 + threadIdx.x;
  if (i < n) p[i] = 0.f;
}
__global__ __launch_bounds__(256) void k_zero_u32(unsigned* p, int n){
  int i = blockIdx.x * 256 + threadIdx.x;
  if (i < n) p[i] = 0u;
}

__device__ __forceinline__ void lds_copy_u4(__bf16* dst, const __bf16* src, int nelem){
  const int nq = nelem >> 3;
  for (int i = threadIdx.x; i < nq; i += 256)
    reinterpret_cast<uint4*>(dst)[i] = reinterpret_cast<const uint4*>(src)[i];
}

// ---------------- 1) history K/V: Kh = x[hist_qid]@W_K + b_K + h1b ; Vh likewise ----------------
__global__ __launch_bounds__(256) void k_hist_kv(
    const __bf16* __restrict__ x_bf, const int* __restrict__ hist_qid,
    const float* __restrict__ b_K, const float* __restrict__ b_V,
    const float* __restrict__ h1_bias, int ch,
    const __bf16* __restrict__ Wk, const __bf16* __restrict__ Wv,
    __bf16* __restrict__ Kh, __bf16* __restrict__ Vh)
{
  __shared__ alignas(16) __bf16 sWk[64*64];
  __shared__ alignas(16) __bf16 sWv[64*64];
  __shared__ alignas(16) __bf16 sA[8][16*64];
  lds_copy_u4(sWk, Wk, 64*64);
  lds_copy_u4(sWv, Wv, 64*64);
  const int lane = threadIdx.x & 31, wv = threadIdx.x >> 5;
  const int tile = blockIdx.x * 8 + wv;
  const size_t row0 = (size_t)tile * 16;
  __bf16* A = sA[wv];
  for (int idx = lane; idx < 16*8; idx += 32){     // 16 rows x 8 uint4 segs (8 bf16 each)
    int r = idx >> 3, seg = idx & 7;
    int qid = hist_qid[row0 + r];
    reinterpret_cast<uint4*>(A + r*64)[seg] =
      reinterpret_cast<const uint4*>(x_bf + (size_t)qid*64)[seg];
  }
  __syncthreads();

  v8f accK[4] = {}, accV[4] = {};
  #pragma unroll
  for (int kb = 0; kb < 2; ++kb){
    v16bf a = frag_A(A, 64, kb*32, lane);
    #pragma unroll
    for (int nb = 0; nb < 4; ++nb){
      accK[nb] = WMMA_BF16(a, frag_B(sWk, kb, nb, lane), accK[nb]);
      accV[nb] = WMMA_BF16(a, frag_B(sWv, kb, nb, lane), accV[nb]);
    }
  }
  const int hl = lane >> 4, n15 = lane & 15;
  #pragma unroll
  for (int nb = 0; nb < 4; ++nb){
    int n = nb*16 + n15;
    float bk = b_K[n] + h1_bias[ch*64 + n];
    float bv = b_V[n];
    #pragma unroll
    for (int r = 0; r < 8; ++r){
      size_t m = row0 + hl*8 + r;
      Kh[m*64 + n] = (__bf16)(accK[nb][r] + bk);
      Vh[m*64 + n] = (__bf16)(accV[nb][r] + bv);
    }
  }
}

// ---------------- 2) per-edge query + masked attention over history ----------------
__global__ __launch_bounds__(256) void k_edge_q_agg(
    const __bf16* __restrict__ x_bf, const __bf16* __restrict__ s_bf,
    const int* __restrict__ e_qid, const int* __restrict__ e_sid,
    const __bf16* __restrict__ Wq, const float* __restrict__ b_Q,
    const __bf16* __restrict__ Kh, const __bf16* __restrict__ Vh,
    const int* __restrict__ hist_cnt, __bf16* __restrict__ agg)
{
  __shared__ alignas(16) __bf16 sWq[128*64];
  __shared__ alignas(16) char   sbuf[8][16*128*2];   // A (bf16) then reused as q (f32)
  __shared__ int sSid[8][16];
  lds_copy_u4(sWq, Wq, 128*64);
  const int lane = threadIdx.x & 31, wv = threadIdx.x >> 5;
  const int tile = blockIdx.x * 8 + wv;
  const size_t e0 = (size_t)tile * 16;
  __bf16* A = reinterpret_cast<__bf16*>(sbuf[wv]);
  for (int idx = lane; idx < 16*16; idx += 32){     // 16 rows x 16 segs: 0-7 x, 8-15 s_emb
    int r = idx >> 4, seg = idx & 15;
    uint4 v;
    if (seg < 8){
      int qid = e_qid[e0 + r];
      v = reinterpret_cast<const uint4*>(x_bf + (size_t)qid*64)[seg];
    } else {
      int sid = e_sid[e0 + r];
      if (seg == 8) sSid[wv][r] = sid;
      v = reinterpret_cast<const uint4*>(s_bf + (size_t)sid*64)[seg - 8];
    }
    reinterpret_cast<uint4*>(A + r*128)[seg] = v;
  }
  __syncthreads();

  v8f acc[4] = {};
  #pragma unroll
  for (int kb = 0; kb < 4; ++kb){
    v16bf a = frag_A(A, 128, kb*32, lane);
    #pragma unroll
    for (int nb = 0; nb < 4; ++nb)
      acc[nb] = WMMA_BF16(a, frag_B(sWq, kb, nb, lane), acc[nb]);
  }
  __syncthreads();
  float* qf = reinterpret_cast<float*>(sbuf[wv]);    // 16x64 f32 q tile, reuses A space
  {
    const int hl = lane >> 4, n15 = lane & 15;
    #pragma unroll
    for (int nb = 0; nb < 4; ++nb){
      float bq = b_Q[nb*16 + n15];
      #pragma unroll
      for (int r = 0; r < 8; ++r)
        qf[(hl*8 + r)*64 + nb*16 + n15] = acc[nb][r] + bq;
    }
  }
  __syncthreads();

  // phase 2: 2 lanes per edge (hl = which half of the 32 history slots)
  const int hl = lane >> 4, e = lane & 15;
  const size_t ge = e0 + e;
  const int sid = sSid[wv][e];
  const int cnt = hist_cnt[sid];
  float qreg[64];
  #pragma unroll
  for (int i = 0; i < 16; ++i){
    float4 t = reinterpret_cast<const float4*>(qf + e*64)[i];
    qreg[4*i] = t.x; qreg[4*i+1] = t.y; qreg[4*i+2] = t.z; qreg[4*i+3] = t.w;
  }

  float aw[16];
  float mx = -1e30f;
  #pragma unroll
  for (int hh = 0; hh < 16; ++hh){
    int h = hl*16 + hh;
    float d = -1e9f;
    if (h < cnt){
      const uint4* kr = reinterpret_cast<const uint4*>(Kh + ((size_t)sid*HN + h)*64);
      float s = 0.f;
      #pragma unroll
      for (int sg = 0; sg < 8; ++sg){
        uint4 u = kr[sg];
        const float* qq = qreg + sg*8;
        s += bflo(u.x)*qq[0] + bfhi(u.x)*qq[1] + bflo(u.y)*qq[2] + bfhi(u.y)*qq[3]
           + bflo(u.z)*qq[4] + bfhi(u.z)*qq[5] + bflo(u.w)*qq[6] + bfhi(u.w)*qq[7];
      }
      d = s * 0.125f;   // 1/sqrt(64)
    }
    aw[hh] = d;
    mx = fmaxf(mx, d);
  }
  mx = fmaxf(mx, __shfl_xor(mx, 16, 32));
  float se = 0.f;
  #pragma unroll
  for (int hh = 0; hh < 16; ++hh){ aw[hh] = __expf(aw[hh] - mx); se += aw[hh]; }
  se += __shfl_xor(se, 16, 32);
  float inv = 1.f / se;
  #pragma unroll
  for (int hh = 0; hh < 16; ++hh)
    aw[hh] = ((hl*16 + hh) < cnt) ? aw[hh] * inv : 0.f;

  #pragma unroll
  for (int fb = 0; fb < 4; ++fb){
    float fa[16];
    #pragma unroll
    for (int j = 0; j < 16; ++j) fa[j] = 0.f;
    for (int hh = 0; hh < 16; ++hh){
      float w = aw[hh];
      if (w == 0.f) continue;
      const uint4* vr = reinterpret_cast<const uint4*>(
          Vh + ((size_t)sid*HN + hl*16 + hh)*64) + fb*2;
      uint4 u0 = vr[0], u1 = vr[1];
      fa[0]+=w*bflo(u0.x); fa[1]+=w*bfhi(u0.x); fa[2]+=w*bflo(u0.y); fa[3]+=w*bfhi(u0.y);
      fa[4]+=w*bflo(u0.z); fa[5]+=w*bfhi(u0.z); fa[6]+=w*bflo(u0.w); fa[7]+=w*bfhi(u0.w);
      fa[8]+=w*bflo(u1.x); fa[9]+=w*bfhi(u1.x); fa[10]+=w*bflo(u1.y); fa[11]+=w*bfhi(u1.y);
      fa[12]+=w*bflo(u1.z); fa[13]+=w*bfhi(u1.z); fa[14]+=w*bflo(u1.w); fa[15]+=w*bfhi(u1.w);
    }
    #pragma unroll
    for (int j = 0; j < 16; ++j) fa[j] += __shfl_xor(fa[j], 16, 32);
    const int jb = hl*8;
    unsigned* dst = reinterpret_cast<unsigned*>(agg + ge*64 + fb*16 + jb);
    #pragma unroll
    for (int t = 0; t < 4; ++t){
      unsigned pk = ((unsigned)bfbits(fa[jb + 2*t + 1]) << 16) | bfbits(fa[jb + 2*t]);
      dst[t] = pk;
    }
  }
}

// ---------------- 3) s_repr = [s_emb[e_sid] || agg] @ W_fuse + b_fuse ----------------
__global__ __launch_bounds__(256) void k_fuse(
    const __bf16* __restrict__ s_bf, const int* __restrict__ e_sid,
    const __bf16* __restrict__ agg, const __bf16* __restrict__ Wf,
    const float* __restrict__ b_fuse, __bf16* __restrict__ srepr)
{
  __shared__ alignas(16) __bf16 sWf[128*64];
  __shared__ alignas(16) __bf16 sA[8][16*128];
  lds_copy_u4(sWf, Wf, 128*64);
  const int lane = threadIdx.x & 31, wv = threadIdx.x >> 5;
  const size_t e0 = (size_t)(blockIdx.x * 8 + wv) * 16;
  __bf16* A = sA[wv];
  for (int idx = lane; idx < 16*16; idx += 32){
    int r = idx >> 4, seg = idx & 15;
    uint4 v;
    if (seg < 8){
      int sid = e_sid[e0 + r];
      v = reinterpret_cast<const uint4*>(s_bf + (size_t)sid*64)[seg];
    } else {
      v = reinterpret_cast<const uint4*>(agg + (e0 + r)*64)[seg - 8];
    }
    reinterpret_cast<uint4*>(A + r*128)[seg] = v;
  }
  __syncthreads();

  v8f acc[4] = {};
  #pragma unroll
  for (int kb = 0; kb < 4; ++kb){
    v16bf a = frag_A(A, 128, kb*32, lane);
    #pragma unroll
    for (int nb = 0; nb < 4; ++nb)
      acc[nb] = WMMA_BF16(a, frag_B(sWf, kb, nb, lane), acc[nb]);
  }
  const int hl = lane >> 4, n15 = lane & 15;
  #pragma unroll
  for (int nb = 0; nb < 4; ++nb){
    int n = nb*16 + n15;
    float bf = b_fuse[n];
    #pragma unroll
    for (int r = 0; r < 8; ++r)
      srepr[(e0 + hl*8 + r)*64 + n] = (__bf16)(acc[nb][r] + bf);
  }
}

// ---------------- 4) sc2 = <x[e_qid], s_repr@W_K2 + b> * scale ; segment max ----------------
__global__ __launch_bounds__(256) void k_k2(
    const __bf16* __restrict__ srepr, const __bf16* __restrict__ x_bf,
    const int* __restrict__ e_qid, const __bf16* __restrict__ Wk2,
    const float* __restrict__ b_K2, const float* __restrict__ h2_bias, int ch,
    float* __restrict__ sc2, unsigned* __restrict__ mx)
{
  __shared__ alignas(16) __bf16 sW[64*64];
  __shared__ alignas(16) __bf16 sA[8][16*64];
  __shared__ int sQid[8][16];
  lds_copy_u4(sW, Wk2, 64*64);
  const int lane = threadIdx.x & 31, wv = threadIdx.x >> 5;
  const size_t e0 = (size_t)(blockIdx.x * 8 + wv) * 16;
  __bf16* A = sA[wv];
  for (int idx = lane; idx < 16*8; idx += 32){
    int r = idx >> 3, seg = idx & 7;
    reinterpret_cast<uint4*>(A + r*64)[seg] =
      reinterpret_cast<const uint4*>(srepr + (e0 + r)*64)[seg];
  }
  if (lane < 16) sQid[wv][lane] = e_qid[e0 + lane];
  __syncthreads();

  v8f acc[4] = {};
  #pragma unroll
  for (int kb = 0; kb < 2; ++kb){
    v16bf a = frag_A(A, 64, kb*32, lane);
    #pragma unroll
    for (int nb = 0; nb < 4; ++nb)
      acc[nb] = WMMA_BF16(a, frag_B(sW, kb, nb, lane), acc[nb]);
  }
  const int hl = lane >> 4, n15 = lane & 15;
  float part[8] = {};
  #pragma unroll
  for (int nb = 0; nb < 4; ++nb){
    int n = nb*16 + n15;
    float bb = b_K2[n] + h2_bias[ch*64 + n];
    #pragma unroll
    for (int r = 0; r < 8; ++r){
      int m = hl*8 + r;
      float xv = (float)x_bf[(size_t)sQid[wv][m]*64 + n];
      part[r] += (acc[nb][r] + bb) * xv;
    }
  }
  #pragma unroll
  for (int r = 0; r < 8; ++r){
    float v = part[r];
    v += __shfl_xor(v, 1, 32); v += __shfl_xor(v, 2, 32);
    v += __shfl_xor(v, 4, 32); v += __shfl_xor(v, 8, 32);
    part[r] = v;
  }
  if (n15 == 0){
    #pragma unroll
    for (int r = 0; r < 8; ++r){
      int m = hl*8 + r;
      float v = part[r] * 0.125f;
      sc2[e0 + m] = v;
      atomicMax(&mx[sQid[wv][m]], fenc(v));
    }
  }
}

// ---------------- 5) ex = exp(sc2 - mx[qid]) ; den[qid] += ex ----------------
__global__ __launch_bounds__(256) void k_expden(
    const int* __restrict__ e_qid, float* __restrict__ scex,
    const unsigned* __restrict__ mx, float* __restrict__ den)
{
  int e = blockIdx.x * 256 + threadIdx.x;
  if (e >= EN) return;
  int q = e_qid[e];
  float ex = __expf(scex[e] - fdec(mx[q]));
  scex[e] = ex;
  atomicAdd(&den[q], ex);
}

// ---------------- 6) out[qid] += alpha * (s_repr@W_V2 + b_V2) ----------------
__global__ __launch_bounds__(256) void k_scatter(
    const __bf16* __restrict__ srepr, const int* __restrict__ e_qid,
    const __bf16* __restrict__ Wv2, const float* __restrict__ b_V2,
    const float* __restrict__ scex, const float* __restrict__ den,
    float* __restrict__ cat, int ch)
{
  __shared__ alignas(16) __bf16 sW[64*64];
  __shared__ alignas(16) __bf16 sA[8][16*64];
  __shared__ int   sQid[8][16];
  __shared__ float sAl[8][16];
  lds_copy_u4(sW, Wv2, 64*64);
  const int lane = threadIdx.x & 31, wv = threadIdx.x >> 5;
  const size_t e0 = (size_t)(blockIdx.x * 8 + wv) * 16;
  __bf16* A = sA[wv];
  for (int idx = lane; idx < 16*8; idx += 32){
    int r = idx >> 3, seg = idx & 7;
    reinterpret_cast<uint4*>(A + r*64)[seg] =
      reinterpret_cast<const uint4*>(srepr + (e0 + r)*64)[seg];
  }
  if (lane < 16){
    int qid = e_qid[e0 + lane];
    sQid[wv][lane] = qid;
    sAl[wv][lane] = scex[e0 + lane] / fmaxf(den[qid], 1e-9f);
  }
  __syncthreads();

  v8f acc[4] = {};
  #pragma unroll
  for (int kb = 0; kb < 2; ++kb){
    v16bf a = frag_A(A, 64, kb*32, lane);
    #pragma unroll
    for (int nb = 0; nb < 4; ++nb)
      acc[nb] = WMMA_BF16(a, frag_B(sW, kb, nb, lane), acc[nb]);
  }
  const int hl = lane >> 4, n15 = lane & 15;
  #pragma unroll
  for (int nb = 0; nb < 4; ++nb){
    int n = nb*16 + n15;
    float bv = b_V2[n];
    #pragma unroll
    for (int r = 0; r < 8; ++r){
      int m = hl*8 + r;
      atomicAdd(&cat[(size_t)sQid[wv][m]*128 + ch*64 + n],
                sAl[wv][m] * (acc[nb][r] + bv));
    }
  }
}

// ---------------- 7) out = cat @ W_proj + b_proj ----------------
__global__ __launch_bounds__(256) void k_proj(
    const float* __restrict__ cat, const __bf16* __restrict__ Wp,
    const float* __restrict__ b_proj, float* __restrict__ out)
{
  __shared__ alignas(16) __bf16 sWp[128*64];
  __shared__ alignas(16) __bf16 sA[8][16*128];
  lds_copy_u4(sWp, Wp, 128*64);
  const int lane = threadIdx.x & 31, wv = threadIdx.x >> 5;
  const size_t r0 = (size_t)(blockIdx.x * 8 + wv) * 16;
  __bf16* A = sA[wv];
  for (int idx = lane; idx < 16*64; idx += 32){      // pack 2 floats -> b32 store
    int r = idx >> 6, c2 = idx & 63;
    float2 t = reinterpret_cast<const float2*>(cat + (r0 + r)*128)[c2];
    unsigned pk = ((unsigned)bfbits(t.y) << 16) | bfbits(t.x);
    reinterpret_cast<unsigned*>(A + r*128)[c2] = pk;
  }
  __syncthreads();

  v8f acc[4] = {};
  #pragma unroll
  for (int kb = 0; kb < 4; ++kb){
    v16bf a = frag_A(A, 128, kb*32, lane);
    #pragma unroll
    for (int nb = 0; nb < 4; ++nb)
      acc[nb] = WMMA_BF16(a, frag_B(sWp, kb, nb, lane), acc[nb]);
  }
  const int hl = lane >> 4, n15 = lane & 15;
  #pragma unroll
  for (int nb = 0; nb < 4; ++nb){
    int n = nb*16 + n15;
    float bp = b_proj[n];
    #pragma unroll
    for (int r = 0; r < 8; ++r)
      out[(r0 + hl*8 + r)*64 + n] = acc[nb][r] + bp;
  }
}

extern "C" void kernel_launch(void* const* d_in, const int* in_sizes, int n_in,
                              void* d_out, int out_size, void* d_ws, size_t ws_size,
                              hipStream_t stream) {
  const float* x      = (const float*)d_in[0];
  const float* s_emb  = (const float*)d_in[1];
  const float* W_Q    = (const float*)d_in[2];  const float* b_Q    = (const float*)d_in[3];
  const float* W_K    = (const float*)d_in[4];  const float* b_K    = (const float*)d_in[5];
  const float* W_V    = (const float*)d_in[6];  const float* b_V    = (const float*)d_in[7];
  const float* W_fuse = (const float*)d_in[8];  const float* b_fuse = (const float*)d_in[9];
  const float* W_K2   = (const float*)d_in[10]; const float* b_K2   = (const float*)d_in[11];
  const float* W_V2   = (const float*)d_in[12]; const float* b_V2   = (const float*)d_in[13];
  const float* W_proj = (const float*)d_in[14]; const float* b_proj = (const float*)d_in[15];
  const float* h1b    = (const float*)d_in[16]; const float* h2b    = (const float*)d_in[17];
  const int* hist_qid[2] = { (const int*)d_in[18], (const int*)d_in[20] };
  const int* hist_cnt[2] = { (const int*)d_in[19], (const int*)d_in[21] };
  const int* e_sid[2]    = { (const int*)d_in[22], (const int*)d_in[24] };
  const int* e_qid[2]    = { (const int*)d_in[23], (const int*)d_in[25] };
  float* out = (float*)d_out;

  size_t off = 0;
  auto alloc = [&](size_t bytes) -> char* {
    char* p = (char*)d_ws + off;
    off += (bytes + 255) & ~(size_t)255;
    return p;
  };
  __bf16* x_bf   = (__bf16*)alloc((size_t)QN*64*2);
  __bf16* s_bf   = (__bf16*)alloc((size_t)SN*64*2);
  __bf16* Wq_sw  = (__bf16*)alloc(128*64*2);
  __bf16* Wk_sw  = (__bf16*)alloc(64*64*2);
  __bf16* Wv_sw  = (__bf16*)alloc(64*64*2);
  __bf16* Wf_sw  = (__bf16*)alloc(128*64*2);
  __bf16* Wk2_sw = (__bf16*)alloc(64*64*2);
  __bf16* Wv2_sw = (__bf16*)alloc(64*64*2);
  __bf16* Wp_sw  = (__bf16*)alloc(128*64*2);
  __bf16* Kh     = (__bf16*)alloc((size_t)SN*HN*64*2);
  __bf16* Vh     = (__bf16*)alloc((size_t)SN*HN*64*2);
  __bf16* agg    = (__bf16*)alloc((size_t)EN*64*2);
  __bf16* srepr  = (__bf16*)alloc((size_t)EN*64*2);
  float*  scex   = (float*) alloc((size_t)EN*4);
  unsigned* mx   = (unsigned*)alloc((size_t)QN*4);
  float*  den    = (float*) alloc((size_t)QN*4);
  float*  cat    = (float*) alloc((size_t)QN*128*4);

  k_cvt_bf16<<<(QN*64 + 255)/256, 256, 0, stream>>>(x, x_bf, QN*64);
  k_cvt_bf16<<<(SN*64 + 255)/256, 256, 0, stream>>>(s_emb, s_bf, SN*64);
  auto swz = [&](const float* W, __bf16* d, int K){
    k_swz_w<<<(K*64 + 255)/256, 256, 0, stream>>>(W, d, K);
  };
  swz(W_Q, Wq_sw, 128);  swz(W_K, Wk_sw, 64);   swz(W_V, Wv_sw, 64);
  swz(W_fuse, Wf_sw, 128); swz(W_K2, Wk2_sw, 64); swz(W_V2, Wv2_sw, 64);
  swz(W_proj, Wp_sw, 128);

  k_zero_f32<<<(QN*128 + 255)/256, 256, 0, stream>>>(cat, QN*128);

  const int gHist = (SN*HN/16)/8;   // 4096
  const int gEdge = (EN/16)/8;      // 3072
  for (int ch = 0; ch < 2; ++ch){
    k_zero_u32<<<(QN + 255)/256, 256, 0, stream>>>(mx, QN);
    k_zero_f32<<<(QN + 255)/256, 256, 0, stream>>>(den, QN);

    k_hist_kv<<<gHist, 256, 0, stream>>>(x_bf, hist_qid[ch], b_K, b_V, h1b, ch,
                                         Wk_sw, Wv_sw, Kh, Vh);
    k_edge_q_agg<<<gEdge, 256, 0, stream>>>(x_bf, s_bf, e_qid[ch], e_sid[ch],
                                            Wq_sw, b_Q, Kh, Vh, hist_cnt[ch], agg);
    k_fuse<<<gEdge, 256, 0, stream>>>(s_bf, e_sid[ch], agg, Wf_sw, b_fuse, srepr);
    k_k2<<<gEdge, 256, 0, stream>>>(srepr, x_bf, e_qid[ch], Wk2_sw, b_K2, h2b, ch,
                                    scex, mx);
    k_expden<<<(EN + 255)/256, 256, 0, stream>>>(e_qid[ch], scex, mx, den);
    k_scatter<<<gEdge, 256, 0, stream>>>(srepr, e_qid[ch], Wv2_sw, b_V2,
                                         scex, den, cat, ch);
  }
  k_proj<<<(QN/16)/8, 256, 0, stream>>>(cat, Wp_sw, b_proj, out);
}
